// MultiHeadSelfAttention_17480516895364
// MI455X (gfx1250) — matrix operand order repaired and checked by reading, not compiled, not run on hardware
//
#include <hip/hip_runtime.h>
#include <hip/hip_bf16.h>

// ---------------------------------------------------------------------------
// MHA forward for MI455X (gfx1250, wave32, WMMA + TDM).
// fp32 inputs -> bf16 WMMA (v_wmma_f32_16x16x32_bf16) with fp32 accumulate.
// Projection GEMMs: 64x64 block tile, B staged in LDS by the Tensor Data
// Mover (tensor_load_to_lds, double buffered, TENSORcnt-synchronized).
// Attention: flash-style streaming softmax (no 268MB score tensor).
// ---------------------------------------------------------------------------

typedef __attribute__((ext_vector_type(16))) __bf16 v16bf;
typedef __attribute__((ext_vector_type(8)))  __bf16 v8bf;
typedef __attribute__((ext_vector_type(8)))  float  v8f;
typedef __attribute__((ext_vector_type(4)))  unsigned int u32x4;
typedef __attribute__((ext_vector_type(8)))  int i32x8;
typedef __attribute__((ext_vector_type(4)))  int i32x4;

#define BATCH 2
#define SEQ   2048
#define EMB   1024
#define HEADS 16
#define DH    64
#define MROWS (BATCH * SEQ)   // 4096

union BF16x16 { v16bf v; v8bf h[2]; };

__device__ __forceinline__ v16bf load_frag(const __bf16* p0, const __bf16* p1) {
  BF16x16 u;
  u.h[0] = *(const v8bf*)p0;   // 16B aligned by construction
  u.h[1] = *(const v8bf*)p1;
  return u.v;
}

__device__ __forceinline__ v8f wmma_bf16(v16bf a, v16bf b, v8f c) {
  return __builtin_amdgcn_wmma_f32_16x16x32_bf16(
      /*neg_a=*/false, a, /*neg_b=*/false, b,
      /*c_mod=*/(short)0, c, /*reuse_a=*/false, /*reuse_b=*/false);
}

// LDS byte offset of a __shared__ object (flat LDS aperture: addr[31:0] = offset)
__device__ __forceinline__ unsigned lds_off_of(const void* p) {
  return (unsigned)(uintptr_t)p;
}

// ---------------------------------------------------------------------------
// TDM: issue a 2-D tile load (rows x cols of bf16) from a row-major tensor
// into LDS. D# bit layout per CDNA5 ISA 8.3/8.4 (groups 0/1; groups 2/3 null).
// ---------------------------------------------------------------------------
__device__ __forceinline__ void tdm_load_tile_2d(
    unsigned lds_byte_off, const __bf16* gptr,
    unsigned tensor_cols, unsigned tensor_rows, unsigned stride_elems,
    unsigned tile_cols, unsigned tile_rows) {
  unsigned long long ga = (unsigned long long)(uintptr_t)gptr;
  u32x4 g0;
  g0[0] = 1u;                                   // count=1, user-mode load D#
  g0[1] = lds_byte_off;                         // lds_addr [63:32]
  g0[2] = (unsigned)(ga & 0xFFFFFFFFu);         // global_addr[31:0]
  g0[3] = (unsigned)((ga >> 32) & 0x01FFFFFFu)  // global_addr[56:32]
        | (2u << 30);                           // type=2 ("image")
  i32x8 g1;
  g1[0] = (int)(1u << 16);                      // workgroup_mask=0, data_size=1 (2B)
  g1[1] = (int)((tensor_cols & 0xFFFFu) << 16); // tensor_dim0[15:0] @ bits 63:48
  g1[2] = (int)(((tensor_cols >> 16) & 0xFFFFu)
        | ((tensor_rows & 0xFFFFu) << 16));     // dim0 hi | dim1 lo
  g1[3] = (int)(((tensor_rows >> 16) & 0xFFFFu)
        | ((tile_cols & 0xFFFFu) << 16));       // dim1 hi | tile_dim0
  g1[4] = (int)(tile_rows & 0xFFFFu);           // tile_dim1 | tile_dim2=0
  g1[5] = (int)stride_elems;                    // tensor_dim0_stride[31:0]
  g1[6] = 0;                                    // stride hi, dim1_stride lo
  g1[7] = 0;
  i32x4 gz4 = {0, 0, 0, 0};
  i32x8 gz8 = {0, 0, 0, 0, 0, 0, 0, 0};
  // clang-23 / therock-10.0 signature: 6 args (extra int32x8 before cpol)
  __builtin_amdgcn_tensor_load_to_lds(g0, g1, gz4, gz4, gz8, 0);
}

// ---------------------------------------------------------------------------
// fp32 -> bf16 conversion (x and the four weight matrices)
// ---------------------------------------------------------------------------
__global__ void f32_to_bf16_kernel(const float* __restrict__ in,
                                   __bf16* __restrict__ out, int n) {
  int i = blockIdx.x * blockDim.x + threadIdx.x;
  if (i < n) out[i] = (__bf16)in[i];
}

// ---------------------------------------------------------------------------
// GEMM: Out[m,n] = sum_k A[m,k] * W[n,k] + bias[n]
//   A: [M,K] bf16 row-major, W: [N,K] bf16 row-major (contraction contiguous).
//   Block = 128 threads (4 waves) computing a 64(M) x 64(N) tile.
//   B tile (64 rows x 32 k) double-buffered in LDS via TDM; each wave reuses
//   one A-fragment across the 4 N-subtiles (4 WMMAs per k-chunk per wave).
//   mode 0: write bf16 head-split  [B,H,S,Dh]   (Q, K projections)
//   mode 1: write bf16 transposed  [B,H,Dh,S]   (V projection -> Vt)
//   mode 2: write fp32 row-major   [M,N]        (final output projection)
// ---------------------------------------------------------------------------
__global__ __launch_bounds__(128)
void gemm64_bf16(const __bf16* __restrict__ A, const __bf16* __restrict__ W,
                 const float* __restrict__ bias, void* __restrict__ Out,
                 int M, int N, int K, int mode) {
  __shared__ __bf16 Bl[2][64][32];              // 2 x 4KB B tiles
  const int lane = threadIdx.x & 31;
  const int wave = threadIdx.x >> 5;            // 0..3
  const int sub  = lane >> 4;
  const int r    = lane & 15;
  const int m0   = blockIdx.x * 64;             // block M origin
  const int n0   = blockIdx.y * 64;             // block N origin

  const __bf16* arow = A + (size_t)(m0 + wave * 16 + r) * K;
  const __bf16* wbase = W + (size_t)n0 * K;

  const int nchunks = K / 32;
  if (wave == 0)
    tdm_load_tile_2d(lds_off_of(&Bl[0][0][0]), wbase,
                     (unsigned)K, (unsigned)N, (unsigned)K, 32u, 64u);

  v8f c[4] = {};                                // stays in VGPRs (no &-taking)
  for (int ch = 0; ch < nchunks; ++ch) {
    const int cur = ch & 1;
    if (wave == 0) __builtin_amdgcn_s_wait_tensorcnt(0);
    __syncthreads();                            // Bl[cur] ready for all waves
    if (wave == 0 && ch + 1 < nchunks)          // overlap next TDM with compute
      tdm_load_tile_2d(lds_off_of(&Bl[cur ^ 1][0][0]),
                       wbase + (size_t)(ch + 1) * 32,
                       (unsigned)K, (unsigned)N, (unsigned)K, 32u, 64u);

    const int kc = ch * 32;
    // A-frag (16x32): lane row, K = {kc+8s..+7, kc+16+8s..+7}; reused x4
    v16bf af = load_frag(arow + kc + 8 * sub, arow + kc + 16 + 8 * sub);
    if (kc + 64 < K)
      __builtin_prefetch(arow + kc + 64, 0, 3); // WGP-scope global_prefetch_b8
    // B-frags from LDS: lane col = t*16 + r, K = {8*sub..+7, 16+8*sub..+7}
#pragma unroll
    for (int t = 0; t < 4; ++t) {
      const __bf16* bt = &Bl[cur][t * 16 + r][8 * sub];
      c[t] = wmma_bf16(af, load_frag(bt, bt + 16), c[t]);
    }
  }

  // C layout: reg j on this lane = (M = m0 + wave*16 + j + 8*sub, N = n0 + t*16 + r)
#pragma unroll
  for (int t = 0; t < 4; ++t) {
    const int n = n0 + t * 16 + r;
    const float bn = bias[n];
    const int h = n >> 6;                       // DH = 64
    const int d = n & 63;
#pragma unroll
    for (int j = 0; j < 8; ++j) {
      const int m = m0 + wave * 16 + j + 8 * sub;
      const float v = c[t][j] + bn;
      if (mode == 2) {
        ((float*)Out)[(size_t)m * N + n] = v;
      } else {
        const int b = m >> 11;                  // SEQ = 2048
        const int s = m & 2047;
        if (mode == 0)                          // [B*H, S, DH]
          ((__bf16*)Out)[(((size_t)(b * HEADS + h) * SEQ) + s) * DH + d] = (__bf16)v;
        else                                    // mode 1: Vt [B*H, DH, S]
          ((__bf16*)Out)[(((size_t)(b * HEADS + h) * DH) + d) * SEQ + s] = (__bf16)v;
      }
    }
  }
}

// ---------------------------------------------------------------------------
// Flash attention: one wave per (b,h, 16-query tile).
// Per 32-key chunk: 4 WMMAs for E = Q*K^T (16x32), online softmax update,
// P -> A-layout via LDS bounce, 4 WMMAs for O += P*Vchunk (16x64).
// ---------------------------------------------------------------------------
__global__ __launch_bounds__(32)
void flash_attn_kernel(const __bf16* __restrict__ Qb,
                       const __bf16* __restrict__ Kb,
                       const __bf16* __restrict__ Vt,
                       __bf16* __restrict__ Ob /* [B*S, EMB] bf16 */) {
  __shared__ float Pl[16][33];     // 16x32 P tile, padded row
  const int lane = threadIdx.x;
  const int sub  = lane >> 4;
  const int r    = lane & 15;
  const int bh   = blockIdx.x;       // 0..31 = b*16 + h
  const int q0   = blockIdx.y * 16;

  const __bf16* Qbase = Qb + ((size_t)bh * SEQ + q0) * DH;
  const __bf16* Kbase = Kb + (size_t)bh * SEQ * DH;
  const __bf16* Vbase = Vt + (size_t)bh * DH * SEQ;

  // Q A-fragments for d-chunks {0..31} and {32..63}, loaded once.
  const __bf16* qrow = Qbase + (size_t)r * DH;
  v16bf qf0 = load_frag(qrow + 0  + 8 * sub, qrow + 16 + 8 * sub);
  v16bf qf1 = load_frag(qrow + 32 + 8 * sub, qrow + 48 + 8 * sub);

  float m_run[8], l_run[8];
  v8f o[4] = {};
#pragma unroll
  for (int j = 0; j < 8; ++j) { m_run[j] = -1e30f; l_run[j] = 0.f; }
  const float scale = 0.125f;      // 1/sqrt(DH)

  for (int kc = 0; kc < SEQ; kc += 32) {
    // ---- E = Q * K^T for keys [kc, kc+32): two 16x16 tiles -------------
    v8f e0 = {}, e1 = {};
    const __bf16* krow0 = Kbase + (size_t)(kc + r) * DH;        // keys kc+r
    const __bf16* krow1 = Kbase + (size_t)(kc + 16 + r) * DH;   // keys kc+16+r
    {
      v16bf b00 = load_frag(krow0 + 0  + 8 * sub, krow0 + 16 + 8 * sub);
      v16bf b10 = load_frag(krow1 + 0  + 8 * sub, krow1 + 16 + 8 * sub);
      e0 = wmma_bf16(qf0, b00, e0);
      e1 = wmma_bf16(qf0, b10, e1);
      v16bf b01 = load_frag(krow0 + 32 + 8 * sub, krow0 + 48 + 8 * sub);
      v16bf b11 = load_frag(krow1 + 32 + 8 * sub, krow1 + 48 + 8 * sub);
      e0 = wmma_bf16(qf1, b01, e0);
      e1 = wmma_bf16(qf1, b11, e1);
    }

    // ---- online softmax: rowmax over 32 keys ---------------------------
    float mx[8];
#pragma unroll
    for (int j = 0; j < 8; ++j) {
      e0[j] *= scale; e1[j] *= scale;
      mx[j] = fmaxf(e0[j], e1[j]);
    }
#pragma unroll
    for (int off = 1; off < 16; off <<= 1)
#pragma unroll
      for (int j = 0; j < 8; ++j)
        mx[j] = fmaxf(mx[j], __shfl_xor(mx[j], off, 32));

    float alpha[8], ls[8];
#pragma unroll
    for (int j = 0; j < 8; ++j) {
      const float nm = fmaxf(m_run[j], mx[j]);
      alpha[j] = __expf(m_run[j] - nm);
      m_run[j] = nm;
      e0[j] = __expf(e0[j] - nm);
      e1[j] = __expf(e1[j] - nm);
      ls[j] = e0[j] + e1[j];
    }
#pragma unroll
    for (int off = 1; off < 16; off <<= 1)
#pragma unroll
      for (int j = 0; j < 8; ++j)
        ls[j] += __shfl_xor(ls[j], off, 32);
#pragma unroll
    for (int j = 0; j < 8; ++j)
      l_run[j] = l_run[j] * alpha[j] + ls[j];
#pragma unroll
    for (int t = 0; t < 4; ++t)
#pragma unroll
      for (int j = 0; j < 8; ++j)
        o[t][j] *= alpha[j];

    // ---- P: C-layout -> LDS -> A-layout (16x32 bf16) -------------------
#pragma unroll
    for (int j = 0; j < 8; ++j) {
      Pl[j + 8 * sub][r]      = e0[j];
      Pl[j + 8 * sub][16 + r] = e1[j];
    }
    asm volatile("s_wait_dscnt 0" ::: "memory");  // store->load same-wave fence
    BF16x16 u;
    {
      const float* prow = &Pl[r][0];
      const int base0 = 8 * sub;        // local k {0..7} or {8..15}
      const int base1 = 16 + 8 * sub;   // local k {16..23} or {24..31}
#pragma unroll
      for (int i = 0; i < 8; ++i) {
        u.h[0][i] = (__bf16)prow[base0 + i];
        u.h[1][i] = (__bf16)prow[base1 + i];
      }
    }
    asm volatile("" ::: "memory");      // keep LDS reads before next-iter stores

    // ---- O += P * Vchunk: 4 d-subtiles, contraction over 32 keys -------
#pragma unroll
    for (int t = 0; t < 4; ++t) {
      const __bf16* vr = Vbase + (size_t)(t * 16 + r) * SEQ + kc + 8 * sub;
      o[t] = wmma_bf16(u.v, load_frag(vr, vr + 16), o[t]);
    }
  }

  // ---- normalize and write attn output as [B*S, EMB] bf16 --------------
  const int b = bh >> 4, h = bh & 15;
#pragma unroll
  for (int t = 0; t < 4; ++t) {
#pragma unroll
    for (int j = 0; j < 8; ++j) {
      const int s = q0 + j + 8 * sub;
      const size_t m = (size_t)b * SEQ + s;
      const int col = h * DH + t * 16 + r;
      Ob[m * EMB + col] = (__bf16)(o[t][j] / l_run[j]);
    }
  }
}

// ---------------------------------------------------------------------------
// Launcher. Workspace layout (all bf16), ~48 MB total:
//   xb[M*E] | Wq|Wk|Wv|Wo [E*E each] | Qb | Kb | Vt [M*E each] | attn [M*E]
// ---------------------------------------------------------------------------
extern "C" void kernel_launch(void* const* d_in, const int* in_sizes, int n_in,
                              void* d_out, int out_size, void* d_ws, size_t ws_size,
                              hipStream_t stream) {
  const float* x  = (const float*)d_in[0];
  const float* Wq = (const float*)d_in[1];
  const float* bq = (const float*)d_in[2];
  const float* Wk = (const float*)d_in[3];
  const float* bk = (const float*)d_in[4];
  const float* Wv = (const float*)d_in[5];
  const float* bv = (const float*)d_in[6];
  const float* Wo = (const float*)d_in[7];
  const float* bo = (const float*)d_in[8];
  float* out = (float*)d_out;

  const size_t nX = (size_t)MROWS * EMB;   // 4,194,304
  const size_t nW = (size_t)EMB * EMB;     // 1,048,576

  char* p = (char*)d_ws;
  __bf16* xb  = (__bf16*)p; p += nX * 2;
  __bf16* wqb = (__bf16*)p; p += nW * 2;
  __bf16* wkb = (__bf16*)p; p += nW * 2;
  __bf16* wvb = (__bf16*)p; p += nW * 2;
  __bf16* wob = (__bf16*)p; p += nW * 2;
  __bf16* qb  = (__bf16*)p; p += nX * 2;
  __bf16* kb  = (__bf16*)p; p += nX * 2;
  __bf16* vtb = (__bf16*)p; p += nX * 2;
  __bf16* ab  = (__bf16*)p; p += nX * 2;

  // 1) fp32 -> bf16 conversions
  f32_to_bf16_kernel<<<(int)((nX + 255) / 256), 256, 0, stream>>>(x,  xb,  (int)nX);
  f32_to_bf16_kernel<<<(int)((nW + 255) / 256), 256, 0, stream>>>(Wq, wqb, (int)nW);
  f32_to_bf16_kernel<<<(int)((nW + 255) / 256), 256, 0, stream>>>(Wk, wkb, (int)nW);
  f32_to_bf16_kernel<<<(int)((nW + 255) / 256), 256, 0, stream>>>(Wv, wvb, (int)nW);
  f32_to_bf16_kernel<<<(int)((nW + 255) / 256), 256, 0, stream>>>(Wo, wob, (int)nW);

  // 2) Q/K/V projections (WMMA + TDM), head-split; V written transposed
  dim3 ggrid(MROWS / 64, EMB / 64);
  gemm64_bf16<<<ggrid, 128, 0, stream>>>(xb, wqb, bq, qb,  MROWS, EMB, EMB, 0);
  gemm64_bf16<<<ggrid, 128, 0, stream>>>(xb, wkb, bk, kb,  MROWS, EMB, EMB, 0);
  gemm64_bf16<<<ggrid, 128, 0, stream>>>(xb, wvb, bv, vtb, MROWS, EMB, EMB, 1);

  // 3) Flash attention (WMMA), writes [B*S, EMB] bf16
  flash_attn_kernel<<<dim3(BATCH * HEADS, SEQ / 16), 32, 0, stream>>>(qb, kb, vtb, ab);

  // 4) Output projection (WMMA + TDM), fp32 result
  gemm64_bf16<<<ggrid, 128, 0, stream>>>(ab, wob, bo, out, MROWS, EMB, EMB, 2);
}
